// GraphAttentionConv_45887430590542
// MI455X (gfx1250) — matrix operand order repaired and checked by reading, not compiled, not run on hardware
//
#include <hip/hip_runtime.h>

// ---------------------------------------------------------------------------
// GraphAttentionConv (GAT w/ edge features) for MI455X (gfx1250, wave32, WMMA)
//
// Roofline: ~1 GB traffic @ 23.3 TB/s ~= 45us; 52 GFLOP of edge GEMMs is
// negligible on the f16 WMMA path (v_wmma_f32_16x16x32_f16, fp32 accum), so
// we convert operands to f16 and make the kernel HBM-bound.
// Q is computed per-NODE (x[dst]@Wq == Q[dst]) -- 16x less GEMM work than the
// naive per-edge formulation.  K/V stay per-edge due to edge_attr.
// ---------------------------------------------------------------------------

typedef __attribute__((ext_vector_type(16))) _Float16 v16h;
typedef __attribute__((ext_vector_type(8)))  _Float16 v8h;
typedef __attribute__((ext_vector_type(8)))  float    v8f;

#define N_NODES 50000
#define N_EDGES 800000
#define HID     128
#define HEADS   4
#define LDSW    144   // padded LDS row stride in halves: 288B = 9*32B -> v16h loads stay 32B aligned

// ---------------- small helpers ----------------

__device__ __forceinline__ void atomic_max_float(float* addr, float val) {
  // int-punning float max: correct for mixed signs, init must be -inf
  if (val >= 0.0f) atomicMax((int*)addr, __float_as_int(val));
  else             atomicMin((unsigned int*)addr, __float_as_uint(val));
}

// stage a 128x128 f16 weight matrix (n-major) into padded LDS
__device__ __forceinline__ void stage_weights(const _Float16* __restrict__ wt,
                                              _Float16* lds) {
  for (int c = threadIdx.x; c < 128 * 16; c += blockDim.x) {
    int n  = c >> 4;
    int ko = (c & 15) << 3;
    *(v8h*)(lds + n * LDSW + ko) = *(const v8h*)(wt + n * 128 + ko);
  }
  __syncthreads();
}

// A-tile (16x32 f16) from f16 node features, row-major, stride 128
__device__ __forceinline__ v16h load_a_node(const _Float16* __restrict__ xh,
                                            int row, int kt, int h) {
  const _Float16* p = xh + (size_t)row * HID + kt * 32 + h * 16;
  v8h lo = *(const v8h*)p;          // K = 32kt + 16h + [0,8)
  v8h hi = *(const v8h*)(p + 8);    // K = 32kt + 16h + 8 + [0,8)
  return __builtin_shufflevector(lo, hi, 0,1,2,3,4,5,6,7,8,9,10,11,12,13,14,15);
}

// A-tile for edges: f16(x[src] + edge_attr[e])
__device__ __forceinline__ v16h load_a_edge(const _Float16* __restrict__ xh,
                                            const float* __restrict__ ea_row,
                                            long long src, int kt, int h) {
  int k0 = kt * 32 + h * 16;
  v8h xlo = *(const v8h*)(xh + (size_t)src * HID + k0);
  v8h xhi = *(const v8h*)(xh + (size_t)src * HID + k0 + 8);
  v8f elo = *(const v8f*)(ea_row + k0);
  v8f ehi = *(const v8f*)(ea_row + k0 + 8);
  v16h a;
#pragma unroll
  for (int j = 0; j < 8; ++j) {
    a[j]     = (_Float16)((float)xlo[j] + elo[j]);
    a[j + 8] = (_Float16)((float)xhi[j] + ehi[j]);
  }
  return a;
}

// ---------------- prep kernels ----------------

// wt[n][k] = (f16) w[k][n]  (so B-tiles read contiguous K per lane)
__global__ void k_prep_w(const float* __restrict__ w, _Float16* __restrict__ wt) {
  int i = blockIdx.x * blockDim.x + threadIdx.x;
  if (i >= HID * HID) return;
  int n = i >> 7, k = i & 127;
  wt[i] = (_Float16)w[k * HID + n];
}

__global__ void k_convert_x(const float* __restrict__ x, _Float16* __restrict__ xh) {
  int i = blockIdx.x * blockDim.x + threadIdx.x;
  if (i < N_NODES * HID) xh[i] = (_Float16)x[i];
}

__global__ void k_init(float* __restrict__ out, float* __restrict__ seg_max,
                       float* __restrict__ seg_sum) {
  int i = blockIdx.x * blockDim.x + threadIdx.x;
  if (i < N_NODES * HID) out[i] = 0.0f;
  if (i < N_NODES * HEADS) {
    seg_max[i] = -__builtin_huge_valf();
    seg_sum[i] = 0.0f;
  }
}

// ---------------- node GEMM: Q = x @ Wq + bq (f16 out) ----------------

__global__ void __launch_bounds__(128)
k_node_q(const _Float16* __restrict__ xh, const _Float16* __restrict__ wqt,
         const float* __restrict__ bq, _Float16* __restrict__ qh) {
  __shared__ _Float16 lds[128 * LDSW];
  stage_weights(wqt, lds);

  int wave = threadIdx.x >> 5;
  int tile = blockIdx.x * 4 + wave;           // 16 rows per wave
  if (tile >= N_NODES / 16) return;
  int lane = threadIdx.x & 31;
  int mrow = lane & 15, h = lane >> 4;
  int r0 = tile * 16;

  v8f acc[8] = {};
#pragma unroll
  for (int kt = 0; kt < 4; ++kt) {
    v16h a = load_a_node(xh, r0 + mrow, kt, h);
#pragma unroll
    for (int t = 0; t < 8; ++t) {
      v16h b = *(const v16h*)(lds + (t * 16 + mrow) * LDSW + kt * 32 + h * 16);
      acc[t] = __builtin_amdgcn_wmma_f32_16x16x32_f16(false, a, false, b,
                                                      (short)0, acc[t], false, false);
    }
  }
#pragma unroll
  for (int t = 0; t < 8; ++t) {
    int n = t * 16 + mrow;
    float bias = bq[n];
#pragma unroll
    for (int r = 0; r < 8; ++r) {
      int m = r + 8 * h;                       // C/D layout: lane holds col n, rows r+8h
      qh[(size_t)(r0 + m) * HID + n] = (_Float16)(acc[t][r] + bias);
    }
  }
}

// ------- edge GEMM (K) + attention logits + segment max -------

__global__ void __launch_bounds__(128)
k_edge_logits(const _Float16* __restrict__ xh, const float* __restrict__ ea,
              const long long* __restrict__ eidx, const _Float16* __restrict__ wkt,
              const float* __restrict__ bk, const _Float16* __restrict__ qh,
              float* __restrict__ logits, float* __restrict__ seg_max) {
  __shared__ _Float16 lds[128 * LDSW];
  stage_weights(wkt, lds);

  int wave = threadIdx.x >> 5;
  int tile = blockIdx.x * 4 + wave;            // E/16 tiles, grid exact
  int lane = threadIdx.x & 31;
  int mrow = lane & 15, h = lane >> 4;
  int base = tile * 16;

  const long long* src = eidx;
  const long long* dst = eidx + N_EDGES;

  long long s_m = src[base + mrow];
  const float* ea_row = ea + (size_t)(base + mrow) * HID;

  long long dsts[8];
#pragma unroll
  for (int r = 0; r < 8; ++r) dsts[r] = dst[base + r + 8 * h];

  v8f acc[8] = {};
#pragma unroll
  for (int kt = 0; kt < 4; ++kt) {
    if (kt < 3) __builtin_prefetch(ea_row + (kt + 1) * 32, 0, 1);
    v16h a = load_a_edge(xh, ea_row, s_m, kt, h);
#pragma unroll
    for (int t = 0; t < 8; ++t) {
      v16h b = *(const v16h*)(lds + (t * 16 + mrow) * LDSW + kt * 32 + h * 16);
      acc[t] = __builtin_amdgcn_wmma_f32_16x16x32_f16(false, a, false, b,
                                                      (short)0, acc[t], false, false);
    }
  }
#pragma unroll
  for (int t = 0; t < 8; ++t) {
    float bias = bk[t * 16 + mrow];
#pragma unroll
    for (int r = 0; r < 8; ++r) acc[t][r] += bias;
  }

  const float scale = 0.17677669529663687f;    // 1/sqrt(32)
#pragma unroll
  for (int hd = 0; hd < HEADS; ++hd) {
    float lp[8];
#pragma unroll
    for (int r = 0; r < 8; ++r) lp[r] = 0.0f;
#pragma unroll
    for (int tt = 0; tt < 2; ++tt) {           // each head spans two 16-col tiles
      int t = hd * 2 + tt;
      int n = t * 16 + mrow;
#pragma unroll
      for (int r = 0; r < 8; ++r) {
        float qv = (float)qh[(size_t)dsts[r] * HID + n];
        lp[r] += acc[t][r] * qv;
      }
    }
#pragma unroll
    for (int r = 0; r < 8; ++r)
      for (int off = 1; off < 16; off <<= 1)
        lp[r] += __shfl_xor(lp[r], off, 32);   // reduce over the 16 columns
    if (mrow == 0) {
#pragma unroll
      for (int r = 0; r < 8; ++r) {
        int e = base + r + 8 * h;
        float lg = lp[r] * scale;
        logits[e * 4 + hd] = lg;
        atomic_max_float(&seg_max[(int)dsts[r] * 4 + hd], lg);
      }
    }
  }
}

// ------- exp(logit - segmax) + segment sum -------

__global__ void k_softmax(const long long* __restrict__ eidx, float* __restrict__ logits,
                          const float* __restrict__ seg_max, float* __restrict__ seg_sum) {
  int i = blockIdx.x * blockDim.x + threadIdx.x;
  if (i >= N_EDGES * HEADS) return;
  int e = i >> 2, hd = i & 3;
  long long d = eidx[N_EDGES + e];
  float m = seg_max[(int)d * 4 + hd];
  if (!(m > -1e38f && m < 1e38f)) m = 0.0f;    // reference's isfinite guard
  float ex = __expf(logits[i] - m);
  logits[i] = ex;                               // reuse buffer for exp values
  atomicAdd(&seg_sum[(int)d * 4 + hd], ex);
}

// ------- edge GEMM (V) + alpha*v scatter-add -------

__global__ void __launch_bounds__(128)
k_edge_agg(const _Float16* __restrict__ xh, const float* __restrict__ ea,
           const long long* __restrict__ eidx, const _Float16* __restrict__ wvt,
           const float* __restrict__ bv, const float* __restrict__ exw,
           const float* __restrict__ seg_sum, float* __restrict__ out) {
  __shared__ _Float16 lds[128 * LDSW];
  stage_weights(wvt, lds);

  int wave = threadIdx.x >> 5;
  int tile = blockIdx.x * 4 + wave;
  int lane = threadIdx.x & 31;
  int mrow = lane & 15, h = lane >> 4;
  int base = tile * 16;

  const long long* src = eidx;
  const long long* dst = eidx + N_EDGES;

  long long s_m = src[base + mrow];
  const float* ea_row = ea + (size_t)(base + mrow) * HID;

  long long dsts[8];
#pragma unroll
  for (int r = 0; r < 8; ++r) dsts[r] = dst[base + r + 8 * h];

  v8f acc[8] = {};
#pragma unroll
  for (int kt = 0; kt < 4; ++kt) {
    if (kt < 3) __builtin_prefetch(ea_row + (kt + 1) * 32, 0, 1);
    v16h a = load_a_edge(xh, ea_row, s_m, kt, h);
#pragma unroll
    for (int t = 0; t < 8; ++t) {
      v16h b = *(const v16h*)(lds + (t * 16 + mrow) * LDSW + kt * 32 + h * 16);
      acc[t] = __builtin_amdgcn_wmma_f32_16x16x32_f16(false, a, false, b,
                                                      (short)0, acc[t], false, false);
    }
  }

#pragma unroll
  for (int hd = 0; hd < HEADS; ++hd) {
    float co[8];
#pragma unroll
    for (int r = 0; r < 8; ++r) {
      int e = base + r + 8 * h;
      co[r] = exw[e * 4 + hd] / (seg_sum[(int)dsts[r] * 4 + hd] + 1e-16f);
    }
#pragma unroll
    for (int tt = 0; tt < 2; ++tt) {
      int t = hd * 2 + tt;
      int n = t * 16 + mrow;
      float bias = bv[n];
#pragma unroll
      for (int r = 0; r < 8; ++r) {
        float val = (acc[t][r] + bias) * co[r];
        atomicAdd(&out[(size_t)dsts[r] * HID + n], val);
      }
    }
  }
}

// ---------------- launch ----------------

extern "C" void kernel_launch(void* const* d_in, const int* in_sizes, int n_in,
                              void* d_out, int out_size, void* d_ws, size_t ws_size,
                              hipStream_t stream) {
  const float*     x    = (const float*)d_in[0];
  const long long* eidx = (const long long*)d_in[1];
  const float*     ea   = (const float*)d_in[2];
  const float*     Wq   = (const float*)d_in[3];
  const float*     bq   = (const float*)d_in[4];
  const float*     Wk   = (const float*)d_in[5];
  const float*     bk   = (const float*)d_in[6];
  const float*     Wv   = (const float*)d_in[7];
  const float*     bv   = (const float*)d_in[8];
  float*           out  = (float*)d_out;

  char* ws = (char*)d_ws;
  size_t off = 0;
  auto alloc = [&](size_t bytes) -> void* {
    void* p = ws + off;
    off = (off + bytes + 255) & ~(size_t)255;
    return p;
  };
  _Float16* wqt     = (_Float16*)alloc((size_t)HID * HID * 2);
  _Float16* wkt     = (_Float16*)alloc((size_t)HID * HID * 2);
  _Float16* wvt     = (_Float16*)alloc((size_t)HID * HID * 2);
  _Float16* xh      = (_Float16*)alloc((size_t)N_NODES * HID * 2);
  _Float16* qh      = (_Float16*)alloc((size_t)N_NODES * HID * 2);
  float*    logits  = (float*)alloc((size_t)N_EDGES * HEADS * 4);
  float*    seg_max = (float*)alloc((size_t)N_NODES * HEADS * 4);
  float*    seg_sum = (float*)alloc((size_t)N_NODES * HEADS * 4);

  // prep: weight transpose+f16, x->f16, init accumulators/output
  k_prep_w<<<(HID * HID + 255) / 256, 256, 0, stream>>>(Wq, wqt);
  k_prep_w<<<(HID * HID + 255) / 256, 256, 0, stream>>>(Wk, wkt);
  k_prep_w<<<(HID * HID + 255) / 256, 256, 0, stream>>>(Wv, wvt);
  k_convert_x<<<(N_NODES * HID + 255) / 256, 256, 0, stream>>>(x, xh);
  k_init<<<(N_NODES * HID + 255) / 256, 256, 0, stream>>>(out, seg_max, seg_sum);

  // per-node Q
  int node_tiles = N_NODES / 16;                      // 3125
  k_node_q<<<(node_tiles + 3) / 4, 128, 0, stream>>>(xh, wqt, bq, qh);

  // per-edge K + logits + segment max
  int edge_blocks = N_EDGES / 64;                     // 12500 (4 waves x 16 edges)
  k_edge_logits<<<edge_blocks, 128, 0, stream>>>(xh, ea, eidx, wkt, bk, qh,
                                                 logits, seg_max);

  // softmax normalize terms
  k_softmax<<<(N_EDGES * HEADS + 255) / 256, 256, 0, stream>>>(eidx, logits,
                                                               seg_max, seg_sum);

  // per-edge V + weighted scatter-add
  k_edge_agg<<<edge_blocks, 128, 0, stream>>>(xh, ea, eidx, wvt, bv, logits,
                                              seg_sum, out);
}
